// EdgeAwareGATBlock_10668698764072
// MI455X (gfx1250) — matrix operand order repaired and checked by reading, not compiled, not run on hardware
//
#include <hip/hip_runtime.h>
#include <hip/hip_bf16.h>
#include <math.h>

#define NN 50000
#define EE 1000000
#define DD 128
#define HH 4
#define CC 32
#define HC 128        // H*C == D (residual works)
#define EDIM 16
#define NEG_SLOPE 0.2f
#define LN_EPS 1e-5f

typedef __attribute__((ext_vector_type(16))) _Float16 v16h;
typedef __attribute__((ext_vector_type(8)))  float    v8f;

// ---- monotonic float <-> uint mapping for atomic segment-max --------------
__device__ __forceinline__ unsigned f2ord(float f) {
  unsigned u = __float_as_uint(f);
  return (u & 0x80000000u) ? ~u : (u | 0x80000000u);
}
__device__ __forceinline__ float ord2f(unsigned u) {
  return (u & 0x80000000u) ? __uint_as_float(u ^ 0x80000000u)
                           : __uint_as_float(~u);
}
__device__ __forceinline__ float lrelu(float t) {
  return t > 0.f ? t : NEG_SLOPE * t;
}

// ---------------------------------------------------------------------------
// Prep A: convert x (f32 row-major) -> f16, fragment-major:
//   dst[(((b*4 + ks)*32 + lane)*16) + h] holds A-fragment half h of lane
//   per ISA 7.12.2 16-bit A 16x32 layout.
// Destination-indexed so writes are fully coalesced.
// ---------------------------------------------------------------------------
__global__ __launch_bounds__(256) void prep_x_frag(
    const float* __restrict__ x, _Float16* __restrict__ xf16)
{
  const int t = blockIdx.x * 256 + threadIdx.x;
  if (t >= NN * DD) return;
  const int h    = t & 15;
  const int lane = (t >> 4) & 31;
  const int ks   = (t >> 9) & 3;
  const int b    = t >> 11;

  const int grp  = lane >> 4;
  const int mrow = lane & 15;
  const int v    = h >> 1;
  const int p    = h & 1;
  const int base = ((v < 4) ? 0 : 16) + grp * 8;
  const int kk   = base + ((v & 3) << 1) + p;
  const int m    = b * 16 + mrow;
  const int k    = ks * 32 + kk;
  xf16[t] = (_Float16)x[(size_t)m * DD + k];
}

// ---------------------------------------------------------------------------
// Prep B: convert Wl,Wr (f32 row-major KxN) -> f16, fragment-major:
//   dst[((((mat*8 + ntile)*4 + ks)*32 + lane)*16) + h]
//   B fragment: K = 8*grp + h + (h>=8 ? 8 : 0), N = ntile*16 + (lane&15)
// ---------------------------------------------------------------------------
__global__ __launch_bounds__(256) void prep_w_frag(
    const float* __restrict__ Wl, const float* __restrict__ Wr,
    _Float16* __restrict__ wf16)
{
  const int t = blockIdx.x * 256 + threadIdx.x;
  if (t >= 2 * DD * HC) return;
  const int h     = t & 15;
  const int lane  = (t >> 4) & 31;
  const int ks    = (t >> 9) & 3;
  const int ntile = (t >> 11) & 7;
  const int mat   = t >> 14;

  const int grp = lane >> 4;
  const int k   = ks * 32 + 8 * grp + h + ((h >= 8) ? 8 : 0);
  const int n   = ntile * 16 + (lane & 15);
  const float s = mat ? Wr[k * HC + n] : Wl[k * HC + n];
  wf16[t] = (_Float16)s;
}

// ---------------------------------------------------------------------------
// Kernel 1: x_l = x@Wl + bl ; x_r = x@Wr + br via v_wmma_f32_16x16x32_f16.
// Fragment-major f16 operands are staged into LDS with CDNA5 async
// global->LDS b128 copies; each fragment load is then 2x ds_load_b128.
// ---------------------------------------------------------------------------
__global__ __launch_bounds__(256) void node_transform_wmma(
    const _Float16* __restrict__ xf16, const _Float16* __restrict__ wf16,
    const float* __restrict__ bl, const float* __restrict__ br,
    float* __restrict__ xl, float* __restrict__ xr)
{
  __shared__ __attribute__((aligned(32))) _Float16 sW[2 * DD * HC]; // 64 KB
  __shared__ __attribute__((aligned(32))) _Float16 sX[16 * DD];     //  4 KB

  const int tid = threadIdx.x;
  const unsigned ldsW = (unsigned)(size_t)&sW[0];
  const unsigned ldsX = (unsigned)(size_t)&sX[0];
  const unsigned long long gW = (unsigned long long)(size_t)wf16;
  const unsigned long long gX =
      (unsigned long long)(size_t)(xf16 + (size_t)blockIdx.x * (16 * DD));

  // async-stage both weight fragment images (64 KB = 4096 x 16B)
  for (int i = tid; i < (2 * DD * HC) / 8; i += 256) {
    asm volatile("global_load_async_to_lds_b128 %0, %1, off"
                 :: "v"(ldsW + (unsigned)i * 16),
                    "v"(gW + (unsigned long long)i * 16)
                 : "memory");
  }
  // async-stage the x tile (4 KB = 256 x 16B)
  asm volatile("global_load_async_to_lds_b128 %0, %1, off"
               :: "v"(ldsX + (unsigned)tid * 16),
                  "v"(gX + (unsigned long long)tid * 16)
               : "memory");
  asm volatile("s_wait_asynccnt 0x0" ::: "memory");
  __syncthreads();

  const int wave = tid >> 5;        // 0..7 -> 16-wide column tile
  const int lane = tid & 31;
  const int grp  = lane >> 4;
  const int lrow = lane & 15;

  const v16h* fX = (const v16h*)sX;
  const v16h* fL = (const v16h*)sW;
  const v16h* fR = (const v16h*)(sW + DD * HC);

  v8f accl = {};
  v8f accr = {};
  #pragma unroll
  for (int ks = 0; ks < 4; ++ks) {
    const v16h a  = fX[ks * 32 + lane];
    const v16h wl = fL[(wave * 4 + ks) * 32 + lane];
    const v16h wr = fR[(wave * 4 + ks) * 32 + lane];
    accl = __builtin_amdgcn_wmma_f32_16x16x32_f16(false, a, false, wl,
                                                  (short)0, accl, false, false);
    accr = __builtin_amdgcn_wmma_f32_16x16x32_f16(false, a, false, wr,
                                                  (short)0, accr, false, false);
  }

  const int ncol = wave * 16 + lrow;
  const int rowBase = blockIdx.x * 16;
  const float blv = bl[ncol], brv = br[ncol];
  #pragma unroll
  for (int r = 0; r < 8; ++r) {
    int m = rowBase + r + 8 * grp;          // C/D layout: M = reg + 8*grp
    xl[(size_t)m * HC + ncol] = accl[r] + blv;
    xr[(size_t)m * HC + ncol] = accr[r] + brv;
  }
}

// ---------------------------------------------------------------------------
// Kernel 0: zero accumulators (d_out), denom; amax to ordered-minimum.
// ---------------------------------------------------------------------------
__global__ __launch_bounds__(256) void init_ws(
    float* __restrict__ out, unsigned* __restrict__ amax_u,
    float* __restrict__ denom)
{
  size_t t = (size_t)blockIdx.x * 256 + threadIdx.x;
  if (t < (size_t)NN * HC) out[t] = 0.f;
  if (t < (size_t)NN * HH) { amax_u[t] = 0u; denom[t] = 0.f; }
}

// ---------------------------------------------------------------------------
// Kernel 2: one wave per edge. Fused edge projection (K=16 VALU FMAs),
// leaky_relu message, per-head <m, att> score, atomic segment-max over dst.
// ---------------------------------------------------------------------------
__global__ __launch_bounds__(256) void edge_alpha(
    const int* __restrict__ ei, const float* __restrict__ ea,
    const float* __restrict__ We, const float* __restrict__ att,
    const float* __restrict__ xl, const float* __restrict__ xr,
    float* __restrict__ alpha, unsigned* __restrict__ amax_u)
{
  const long long e = (long long)blockIdx.x * 8 + (threadIdx.x >> 5);
  if (e >= EE) return;
  const int lane = threadIdx.x & 31;
  const int src = ei[e];
  const int dst = ei[EE + e];
  const int col = lane * 4;

  float ef0 = 0.f, ef1 = 0.f, ef2 = 0.f, ef3 = 0.f;
  #pragma unroll
  for (int k = 0; k < EDIM; ++k) {
    float av = ea[e * EDIM + k];
    const float4 w = *(const float4*)&We[k * HC + col];
    ef0 += av * w.x; ef1 += av * w.y; ef2 += av * w.z; ef3 += av * w.w;
  }
  const float4 xs = *(const float4*)&xl[(size_t)src * HC + col];
  const float4 xd = *(const float4*)&xr[(size_t)dst * HC + col];
  const float4 at = *(const float4*)&att[col];   // (H,C) flat == col index

  float p = lrelu(xs.x + xd.x + ef0) * at.x
          + lrelu(xs.y + xd.y + ef1) * at.y
          + lrelu(xs.z + xd.z + ef2) * at.z
          + lrelu(xs.w + xd.w + ef3) * at.w;

  p += __shfl_xor(p, 1, 32);
  p += __shfl_xor(p, 2, 32);
  p += __shfl_xor(p, 4, 32);

  const int head = lane >> 3;
  if ((lane & 7) == 0) {
    alpha[e * HH + head] = p;
    atomicMax(&amax_u[(size_t)dst * HH + head], f2ord(p));
  }
}

// ---------------------------------------------------------------------------
// Kernel 3: ex = exp(alpha - amax[dst]); denom[dst] += ex (in place)
// ---------------------------------------------------------------------------
__global__ __launch_bounds__(256) void edge_exp_denom(
    const int* __restrict__ ei, const unsigned* __restrict__ amax_u,
    float* __restrict__ alpha, float* __restrict__ denom)
{
  long long t = (long long)blockIdx.x * 256 + threadIdx.x;
  if (t >= (long long)EE * HH) return;
  int e = (int)(t >> 2), h = (int)(t & 3);
  int dst = ei[EE + e];
  float am = ord2f(amax_u[(size_t)dst * HH + h]);
  float ex = __expf(alpha[t] - am);
  alpha[t] = ex;
  atomicAdd(&denom[(size_t)dst * HH + h], ex);
}

// ---------------------------------------------------------------------------
// Kernel 4: one wave per edge: out[dst] += attn * x_l[src]
// ---------------------------------------------------------------------------
__global__ __launch_bounds__(256) void edge_aggregate(
    const int* __restrict__ ei, const float* __restrict__ ex,
    const float* __restrict__ denom, const float* __restrict__ xl,
    float* __restrict__ out)
{
  const long long e = (long long)blockIdx.x * 8 + (threadIdx.x >> 5);
  if (e >= EE) return;
  const int lane = threadIdx.x & 31;
  const int src = ei[e];
  const int dst = ei[EE + e];
  const int head = lane >> 3;
  const float attn = ex[e * HH + head]
                   / (denom[(size_t)dst * HH + head] + 1e-16f);
  const int col = lane * 4;
  const float4 v = *(const float4*)&xl[(size_t)src * HC + col];
  float* o = &out[(size_t)dst * HC + col];
  atomicAdd(o + 0, attn * v.x);
  atomicAdd(o + 1, attn * v.y);
  atomicAdd(o + 2, attn * v.z);
  atomicAdd(o + 3, attn * v.w);
}

// ---------------------------------------------------------------------------
// Kernel 5: one wave per node: +bias, ELU, +x residual, LayerNorm (in place)
// ---------------------------------------------------------------------------
__global__ __launch_bounds__(256) void finalize_ln(
    const float* __restrict__ x, const float* __restrict__ bias_out,
    const float* __restrict__ gamma, const float* __restrict__ beta,
    float* __restrict__ out)
{
  const int node = blockIdx.x * 8 + (threadIdx.x >> 5);
  if (node >= NN) return;
  const int lane = threadIdx.x & 31;
  const int col = lane * 4;

  const float4 o  = *(const float4*)&out[(size_t)node * HC + col];
  const float4 b  = *(const float4*)&bias_out[col];
  const float4 xv = *(const float4*)&x[(size_t)node * HC + col];

  float h[4];
  {
    float v;
    v = o.x + b.x; h[0] = (v > 0.f ? v : __expf(v) - 1.f) + xv.x;
    v = o.y + b.y; h[1] = (v > 0.f ? v : __expf(v) - 1.f) + xv.y;
    v = o.z + b.z; h[2] = (v > 0.f ? v : __expf(v) - 1.f) + xv.z;
    v = o.w + b.w; h[3] = (v > 0.f ? v : __expf(v) - 1.f) + xv.w;
  }

  float s = h[0] + h[1] + h[2] + h[3];
  #pragma unroll
  for (int m = 16; m >= 1; m >>= 1) s += __shfl_xor(s, m, 32);
  const float mu = s * (1.f / HC);

  float d0 = h[0] - mu, d1 = h[1] - mu, d2 = h[2] - mu, d3 = h[3] - mu;
  float vs = d0 * d0 + d1 * d1 + d2 * d2 + d3 * d3;
  #pragma unroll
  for (int m = 16; m >= 1; m >>= 1) vs += __shfl_xor(vs, m, 32);
  const float rstd = rsqrtf(vs * (1.f / HC) + LN_EPS);

  const float4 g  = *(const float4*)&gamma[col];
  const float4 be = *(const float4*)&beta[col];
  float4 y;
  y.x = d0 * rstd * g.x + be.x;
  y.y = d1 * rstd * g.y + be.y;
  y.z = d2 * rstd * g.z + be.z;
  y.w = d3 * rstd * g.w + be.w;
  *(float4*)&out[(size_t)node * HC + col] = y;
}

// ---------------------------------------------------------------------------
extern "C" void kernel_launch(void* const* d_in, const int* in_sizes, int n_in,
                              void* d_out, int out_size, void* d_ws, size_t ws_size,
                              hipStream_t stream) {
  (void)in_sizes; (void)n_in; (void)out_size; (void)ws_size;
  const float* x        = (const float*)d_in[0];
  const int*   ei       = (const int*)d_in[1];
  const float* ea       = (const float*)d_in[2];
  const float* Wl       = (const float*)d_in[3];
  const float* bl       = (const float*)d_in[4];
  const float* Wr       = (const float*)d_in[5];
  const float* br       = (const float*)d_in[6];
  const float* We       = (const float*)d_in[7];
  const float* att      = (const float*)d_in[8];
  const float* bias_out = (const float*)d_in[9];
  const float* gamma    = (const float*)d_in[10];
  const float* beta     = (const float*)d_in[11];
  float* out = (float*)d_out;

  // workspace carve-out (~82 MB)
  float*     xl     = (float*)d_ws;
  float*     xr     = xl + (size_t)NN * HC;
  float*     alpha  = xr + (size_t)NN * HC;            // E*H, reused as ex
  unsigned*  amax_u = (unsigned*)(alpha + (size_t)EE * HH);
  float*     denom  = (float*)(amax_u + (size_t)NN * HH);
  _Float16*  xf16   = (_Float16*)(denom + (size_t)NN * HH);
  _Float16*  wf16   = xf16 + (size_t)NN * DD;

  // 0) zero accumulators
  init_ws<<<(NN * HC + 255) / 256, 256, 0, stream>>>(out, amax_u, denom);

  // prep: fragment-major f16 images of x, Wl, Wr
  prep_w_frag<<<(2 * DD * HC) / 256, 256, 0, stream>>>(Wl, Wr, wf16);
  prep_x_frag<<<(NN * DD) / 256, 256, 0, stream>>>(x, xf16);

  // 1) node transforms via async-LDS staging + WMMA
  node_transform_wmma<<<NN / 16, 256, 0, stream>>>(xf16, wf16, bl, br, xl, xr);

  // 2) edge scores + segment max
  edge_alpha<<<(EE + 7) / 8, 256, 0, stream>>>(ei, ea, We, att, xl, xr,
                                               alpha, amax_u);

  // 3) exp + segment sum
  edge_exp_denom<<<((long long)EE * HH + 255) / 256, 256, 0, stream>>>(
      ei, amax_u, alpha, denom);

  // 4) weighted scatter-aggregation into d_out
  edge_aggregate<<<(EE + 7) / 8, 256, 0, stream>>>(ei, alpha, denom, xl, out);

  // 5) bias + ELU + residual + LayerNorm (in place on d_out)
  finalize_ln<<<(NN + 7) / 8, 256, 0, stream>>>(x, bias_out, gamma, beta, out);
}
